// SimpleMoELayer_54185307407084
// MI455X (gfx1250) — matrix operand order repaired and checked by reading, not compiled, not run on hardware
//
#include <hip/hip_runtime.h>
#include <math.h>

typedef unsigned short u16;
typedef __attribute__((ext_vector_type(16))) __bf16        v16bf;
typedef __attribute__((ext_vector_type(8)))  float         v8f;
typedef __attribute__((ext_vector_type(4)))  unsigned int  u32x4;
typedef __attribute__((ext_vector_type(4)))  float         f32x4;

#define T_TOK 8192
#define H_DIM 1024
#define I_DIM 4096
#define E_NUM 8

__device__ inline u16 f2bf(float f) {
  union { float f; unsigned u; } v; v.f = f;
  unsigned r = v.u + 0x7FFFu + ((v.u >> 16) & 1u);  // round-to-nearest-even
  return (u16)(r >> 16);
}

__device__ inline float gelu_f(float x) {
  const float c0 = 0.7978845608028654f;  // sqrt(2/pi), tanh approximation (jax default)
  return 0.5f * x * (1.0f + tanhf(c0 * (x + 0.044715f * x * x * x)));
}

union FragU { u32x4 q[2]; v16bf v; };

// A fragment (16x32 bf16, M x K): lane m = l&15; elems 0..7 -> K=k0..k0+7,
// elems 8..15 -> K=k0+16..k0+23, with k0 = (lane<16 ? 0 : 8). Row-major src, ld halves.
__device__ inline v16bf load_frag_a(const u16* base, int lane, int ld, int kbase) {
  int m  = lane & 15;
  int k0 = kbase + ((lane & 16) ? 8 : 0);
  const u16* p = base + (size_t)m * ld + k0;
  FragU u;
  u.q[0] = *(const u32x4*)(p);
  u.q[1] = *(const u32x4*)(p + 16);
  return u.v;
}

// B fragment (32x16 bf16, K x N): lane n = l&15; elems 0..7 -> K = kb..kb+7,
// elems 8..15 -> K = kb+16..kb+23, kb = kbase + (lane<16 ? 0 : 8). Source stored N-major: row n, K contiguous, ld halves.
__device__ inline v16bf load_frag_b(const u16* base, int n0, int ld, int kbase, int lane) {
  int n  = lane & 15;
  int kb = kbase + ((lane & 16) ? 8 : 0);
  const u16* p = base + (size_t)(n0 + n) * ld + kb;
  FragU u;
  u.q[0] = *(const u32x4*)(p);
  u.q[1] = *(const u32x4*)(p + 16);
  return u.v;
}

// Software-pipelined 16xN0x(NK*32) GEMM inner loop, depth-4 register double-buffer.
// A from LDS (abase, row-major ld), B from global (bbase, N-major ld).
// Steady-state trip NK-4 is a multiple of 4 -> all A[]/B[] indices constant after
// unroll-4, so the circular buffer stays in VGPRs and waits become partial.
template <int NK>
__device__ inline v8f gemm_rows16(const u16* abase, const u16* bbase, int n0, int ld,
                                  int lane) {
  FragU A[4], B[4];
#pragma unroll
  for (int s = 0; s < 4; ++s) {
    A[s].v = load_frag_a(abase, lane, ld, s * 32);
    B[s].v = load_frag_b(bbase, n0, ld, s * 32, lane);
  }
  v8f acc = {0.f, 0.f, 0.f, 0.f, 0.f, 0.f, 0.f, 0.f};
#pragma unroll 4
  for (int kk = 0; kk < NK - 4; ++kk) {
    acc = __builtin_amdgcn_wmma_f32_16x16x32_bf16(false, A[kk & 3].v, false, B[kk & 3].v,
                                                  (short)0, acc, false, false);
    int pk = (kk + 4) * 32;
    A[kk & 3].v = load_frag_a(abase, lane, ld, pk);
    B[kk & 3].v = load_frag_b(bbase, n0, ld, pk, lane);
  }
#pragma unroll
  for (int kk = NK - 4; kk < NK; ++kk)
    acc = __builtin_amdgcn_wmma_f32_16x16x32_bf16(false, A[kk & 3].v, false, B[kk & 3].v,
                                                  (short)0, acc, false, false);
  return acc;
}

// ---------------------------------------------------------------------------
__global__ void init_kernel(int* cnt, float* psum) {
  int i = threadIdx.x;
  if (i < E_NUM) { cnt[i] = 0; psum[i] = 0.f; }
}

// src f32 [E][R][C] -> dst bf16 [E][C][R]
__global__ void transpose_bf16_kernel(const float* __restrict__ src,
                                      u16* __restrict__ dst, int R, int C) {
  __shared__ u16 tile[32][33];
  size_t eoff = (size_t)blockIdx.z * R * C;
  int c0 = blockIdx.x * 32, r0 = blockIdx.y * 32;
  for (int i = threadIdx.y; i < 32; i += 8)
    tile[i][threadIdx.x] = f2bf(src[eoff + (size_t)(r0 + i) * C + (c0 + threadIdx.x)]);
  __syncthreads();
  for (int i = threadIdx.y; i < 32; i += 8)
    dst[eoff + (size_t)(c0 + i) * R + (r0 + threadIdx.x)] = tile[threadIdx.x][i];
}

// one block per token: logits -> softmax -> top2 -> lists; also x -> bf16
__global__ void router_kernel(const float* __restrict__ x, const float* __restrict__ Wg,
                              const float* __restrict__ bg, u16* __restrict__ xh,
                              int* cnt, int* lst_dst, float* lst_w, float* psum) {
  __shared__ float red[256][E_NUM];
  int t = blockIdx.x, tid = threadIdx.x;
  const float* xr = x + (size_t)t * H_DIM;
  float acc[E_NUM];
#pragma unroll
  for (int e = 0; e < E_NUM; ++e) acc[e] = 0.f;
  for (int h = tid; h < H_DIM; h += 256) {
    float xv = xr[h];
    xh[(size_t)t * H_DIM + h] = f2bf(xv);
    const float* wgr = Wg + (size_t)h * E_NUM;
#pragma unroll
    for (int e = 0; e < E_NUM; ++e) acc[e] += xv * wgr[e];
  }
#pragma unroll
  for (int e = 0; e < E_NUM; ++e) red[tid][e] = acc[e];
  __syncthreads();
  for (int s = 128; s > 0; s >>= 1) {
    if (tid < s) {
#pragma unroll
      for (int e = 0; e < E_NUM; ++e) red[tid][e] += red[tid + s][e];
    }
    __syncthreads();
  }
  if (tid == 0) {
    float p[E_NUM];
    float mx = -1e30f;
#pragma unroll
    for (int e = 0; e < E_NUM; ++e) { p[e] = red[0][e] + bg[e]; mx = fmaxf(mx, p[e]); }
    float sum = 0.f;
#pragma unroll
    for (int e = 0; e < E_NUM; ++e) { p[e] = expf(p[e] - mx); sum += p[e]; }
    float inv = 1.f / sum;
#pragma unroll
    for (int e = 0; e < E_NUM; ++e) p[e] *= inv;
    int i0 = 0;
    for (int e = 1; e < E_NUM; ++e) if (p[e] > p[i0]) i0 = e;          // ties -> lower idx
    int i1 = (i0 == 0) ? 1 : 0;
    for (int e = 0; e < E_NUM; ++e) if (e != i0 && p[e] > p[i1]) i1 = e;
    float w0 = p[i0], w1 = p[i1], s2 = w0 + w1;
    w0 /= s2; w1 /= s2;
    int pos0 = atomicAdd(&cnt[i0], 1);
    lst_dst[i0 * T_TOK + pos0] = t * 2 + 0;
    lst_w[i0 * T_TOK + pos0] = w0;
    int pos1 = atomicAdd(&cnt[i1], 1);
    lst_dst[i1 * T_TOK + pos1] = t * 2 + 1;
    lst_w[i1 * T_TOK + pos1] = w1;
#pragma unroll
    for (int e = 0; e < E_NUM; ++e) atomicAdd(&psum[e], p[e]);
  }
}

// Fused per-expert FFN: y[entry] = gate * (gelu(x W1 + b1) W2 + b2)
// Block = 16 gathered tokens of one expert; 8 waves; dyn LDS:
//   xtile 16x1024 bf16 (32KB) | htile 16x4096 bf16 (128KB) | swgt[16] | sdst[16]
__global__ void __launch_bounds__(256) expert_kernel(
    const u16* __restrict__ xh, const u16* __restrict__ w1t, const u16* __restrict__ w2t,
    const float* __restrict__ b1, const float* __restrict__ b2,
    const int* __restrict__ lst_dst, const float* __restrict__ lst_w,
    const int* __restrict__ cnt, float* __restrict__ yw) {
  extern __shared__ char smem[];
  u16*   xtile = (u16*)smem;                                     // 16 x H_DIM
  u16*   htile = (u16*)(smem + 16 * H_DIM * 2);                  // 16 x I_DIM
  float* swgt  = (float*)(smem + 16 * H_DIM * 2 + 16 * I_DIM * 2);
  int*   sdst  = (int*)(swgt + 16);

  int e = blockIdx.y;
  int tile0 = blockIdx.x * 16;
  int cn = cnt[e];
  if (tile0 >= cn) return;

  int tid = threadIdx.x;
  if (tid < 16) {
    int gi = tile0 + tid;
    int d = 0; float w = 0.f;
    if (gi < cn) { d = lst_dst[e * T_TOK + gi]; w = lst_w[e * T_TOK + gi]; }
    sdst[tid] = d; swgt[tid] = w;
  }
  __syncthreads();

  // gather 16 token rows (bf16) into LDS, 16B chunks
  for (int idx = tid; idx < 16 * (H_DIM / 8); idx += 256) {
    int row = idx >> 7, c4 = idx & 127;
    int tok = sdst[row] >> 1;
    ((u32x4*)xtile)[idx] = ((const u32x4*)(xh + (size_t)tok * H_DIM))[c4];
  }
  __syncthreads();

  int lane = tid & 31, wv = tid >> 5;

  // Phase 1: h = gelu(x @ W1 + b1), waves split the I/16 = 256 N-blocks
  const u16* w1e = w1t + (size_t)e * I_DIM * H_DIM;   // [I][H] bf16
  for (int nb = wv; nb < I_DIM / 16; nb += 8) {
    int n0 = nb * 16;
    v8f acc = gemm_rows16<H_DIM / 32>(xtile, w1e, n0, H_DIM, lane);
    int n = n0 + (lane & 15);
    float bias = b1[e * I_DIM + n];
    int mh = (lane & 16) ? 8 : 0;
#pragma unroll
    for (int r = 0; r < 8; ++r)
      htile[(mh + r) * I_DIM + n] = f2bf(gelu_f(acc[r] + bias));
  }
  __syncthreads();

  // Phase 2: y = h @ W2 + b2, scale by gate, scatter to yw[dst]
  const u16* w2e = w2t + (size_t)e * I_DIM * H_DIM;   // [H][I] bf16
  for (int nb = wv; nb < H_DIM / 16; nb += 8) {
    int n0 = nb * 16;
    v8f acc = gemm_rows16<I_DIM / 32>(htile, w2e, n0, I_DIM, lane);
    int n = n0 + (lane & 15);
    float bias = b2[e * H_DIM + n];
    int mh = (lane & 16) ? 8 : 0;
#pragma unroll
    for (int r = 0; r < 8; ++r) {
      int m = mh + r;
      if (tile0 + m < cn)
        yw[(size_t)sdst[m] * H_DIM + n] = (acc[r] + bias) * swgt[m];
    }
  }
}

__global__ void combine_kernel(const float* __restrict__ x, const float* __restrict__ yw,
                               float* __restrict__ out) {
  size_t i4   = (size_t)blockIdx.x * blockDim.x + threadIdx.x;  // float4 index
  size_t base = i4 * 4;
  size_t tok  = base >> 10;          // / H_DIM
  size_t c4   = (base & 1023) >> 2;
  f32x4 v = ((const f32x4*)x)[i4];
  v = v + ((const f32x4*)(yw + (tok * 2 + 0) * H_DIM))[c4];
  v = v + ((const f32x4*)(yw + (tok * 2 + 1) * H_DIM))[c4];
  ((f32x4*)out)[i4] = v;
}

__global__ void loss_kernel(const int* cnt, const float* psum, float* out_loss) {
  if (threadIdx.x == 0) {
    float s = 0.f;
#pragma unroll
    for (int e = 0; e < E_NUM; ++e)
      s += ((float)cnt[e] / (float)T_TOK) * (psum[e] / (float)T_TOK);
    *out_loss = 0.01f * (float)E_NUM * s;
  }
}

// ---------------------------------------------------------------------------
extern "C" void kernel_launch(void* const* d_in, const int* in_sizes, int n_in,
                              void* d_out, int out_size, void* d_ws, size_t ws_size,
                              hipStream_t stream) {
  const float* x  = (const float*)d_in[0];
  const float* Wg = (const float*)d_in[1];
  const float* bg = (const float*)d_in[2];
  const float* W1 = (const float*)d_in[3];
  const float* b1 = (const float*)d_in[4];
  const float* W2 = (const float*)d_in[5];
  const float* b2 = (const float*)d_in[6];
  float* out = (float*)d_out;

  char* ws = (char*)d_ws;
  u16* xh      = (u16*)ws;  ws += (size_t)T_TOK * H_DIM * 2;
  u16* w1t     = (u16*)ws;  ws += (size_t)E_NUM * I_DIM * H_DIM * 2;
  u16* w2t     = (u16*)ws;  ws += (size_t)E_NUM * I_DIM * H_DIM * 2;
  float* yw    = (float*)ws; ws += (size_t)2 * T_TOK * H_DIM * 4;
  int* lst_dst = (int*)ws;  ws += (size_t)E_NUM * T_TOK * 4;
  float* lst_w = (float*)ws; ws += (size_t)E_NUM * T_TOK * 4;
  int* cnt     = (int*)ws;  ws += 64;
  float* psum  = (float*)ws; ws += 64;

  init_kernel<<<1, 32, 0, stream>>>(cnt, psum);
  transpose_bf16_kernel<<<dim3(I_DIM / 32, H_DIM / 32, E_NUM), dim3(32, 8), 0, stream>>>(
      W1, w1t, H_DIM, I_DIM);
  transpose_bf16_kernel<<<dim3(H_DIM / 32, I_DIM / 32, E_NUM), dim3(32, 8), 0, stream>>>(
      W2, w2t, I_DIM, H_DIM);
  router_kernel<<<T_TOK, 256, 0, stream>>>(x, Wg, bg, xh, cnt, lst_dst, lst_w, psum);

  size_t smem = (size_t)16 * H_DIM * 2 + (size_t)16 * I_DIM * 2 + 16 * 4 + 16 * 4;
  expert_kernel<<<dim3(T_TOK / 16, E_NUM), 256, smem, stream>>>(
      xh, w1t, w2t, b1, b2, lst_dst, lst_w, cnt, yw);

  combine_kernel<<<(T_TOK * H_DIM / 4) / 256, 256, 0, stream>>>(x, yw, out);
  loss_kernel<<<1, 1, 0, stream>>>(cnt, psum, out + ((size_t)out_size - 1));
}
